// HybridBinaryClassifier_65481071403303
// MI455X (gfx1250) — matrix operand order repaired and checked by reading, not compiled, not run on hardware
//
#include <hip/hip_runtime.h>
#include <hip/hip_bf16.h>
#include <math.h>

typedef _Float16 half8  __attribute__((ext_vector_type(8)));
typedef _Float16 half16 __attribute__((ext_vector_type(16)));
typedef float    f32x8  __attribute__((ext_vector_type(8)));

#define B_      128
#define KP      55840        // 55815 padded to multiple of 32
#define KREAL   55815
#define KSTEPS  1745         // KP / 32
#define KSPLIT  16

static __host__ int ceil_div(int a, int b) { return (a + b - 1) / b; }

// ---------------- conv1: (128,3,250,250) -> relu -> (128,6,124,124) ----------------
__global__ void conv1_kernel(const float* __restrict__ x, const float* __restrict__ w,
                             const float* __restrict__ bias, float* __restrict__ out) {
    int idx = blockIdx.x * blockDim.x + threadIdx.x;
    const int HW = 124 * 124;
    const int total = B_ * 6 * HW;
    if (idx >= total) return;
    int ox = idx % 124;
    int oy = (idx / 124) % 124;
    int oc = (idx / HW) % 6;
    int n  = idx / (6 * HW);
    float acc = bias[oc];
    const float* xn = x + (size_t)n * 3 * 250 * 250;
    const float* wc = w + oc * 75;
    for (int ic = 0; ic < 3; ++ic) {
        const float* xc = xn + ic * 62500;
        const float* wk = wc + ic * 25;
        #pragma unroll
        for (int ky = 0; ky < 5; ++ky) {
            int iy = oy * 2 - 1 + ky;
            if (iy < 0 || iy >= 250) continue;
            #pragma unroll
            for (int kx = 0; kx < 5; ++kx) {
                int ix = ox * 2 - 1 + kx;
                if (ix < 0 || ix >= 250) continue;
                acc = fmaf(xc[iy * 250 + ix], wk[ky * 5 + kx], acc);
            }
        }
    }
    out[idx] = fmaxf(acc, 0.0f);
}

// ---------------- maxpool 2x2 stride 1 (generic, f32) ----------------
__global__ void maxpool_kernel(const float* __restrict__ in, float* __restrict__ out,
                               int C, int H, int W, int total) {
    int idx = blockIdx.x * blockDim.x + threadIdx.x;
    if (idx >= total) return;
    int Ho = H - 1, Wo = W - 1;
    int x = idx % Wo;
    int y = (idx / Wo) % Ho;
    int c = (idx / (Wo * Ho)) % C;
    int n = idx / (Wo * Ho * C);
    const float* p = in + (((size_t)n * C + c) * H + y) * W + x;
    float v = fmaxf(fmaxf(p[0], p[1]), fmaxf(p[W], p[W + 1]));
    out[idx] = v;
}

// ---------------- conv2: (128,6,123,123) -> relu -> (128,15,62,62) ----------------
__global__ void conv2_kernel(const float* __restrict__ x, const float* __restrict__ w,
                             const float* __restrict__ bias, float* __restrict__ out) {
    int idx = blockIdx.x * blockDim.x + threadIdx.x;
    const int HW = 62 * 62;
    const int total = B_ * 15 * HW;
    if (idx >= total) return;
    int ox = idx % 62;
    int oy = (idx / 62) % 62;
    int oc = (idx / HW) % 15;
    int n  = idx / (15 * HW);
    float acc = bias[oc];
    const float* xn = x + (size_t)n * 6 * 123 * 123;
    const float* wc = w + oc * 54;
    for (int ic = 0; ic < 6; ++ic) {
        const float* xc = xn + ic * 15129;
        const float* wk = wc + ic * 9;
        #pragma unroll
        for (int ky = 0; ky < 3; ++ky) {
            int iy = oy * 2 - 1 + ky;
            if (iy < 0 || iy >= 123) continue;
            #pragma unroll
            for (int kx = 0; kx < 3; ++kx) {
                int ix = ox * 2 - 1 + kx;
                if (ix < 0 || ix >= 123) continue;
                acc = fmaf(xc[iy * 123 + ix], wk[ky * 3 + kx], acc);
            }
        }
    }
    out[idx] = fmaxf(acc, 0.0f);
}

// ---------------- maxpool2 + f16 convert + K-pad: -> A [128][KP] f16 ----------------
__global__ void pool2_f16_kernel(const float* __restrict__ h2, _Float16* __restrict__ A) {
    int idx = blockIdx.x * blockDim.x + threadIdx.x;
    const int total = B_ * KP;
    if (idx >= total) return;
    int k = idx % KP;
    int n = idx / KP;
    if (k >= KREAL) { A[idx] = (_Float16)0.0f; return; }
    int x = k % 61;
    int y = (k / 61) % 61;
    int c = k / 3721;               // 61*61
    const float* p = h2 + (((size_t)n * 15 + c) * 62 + y) * 62 + x;
    float v = fmaxf(fmaxf(p[0], p[1]), fmaxf(p[62], p[63]));
    A[idx] = (_Float16)v;
}

// ---------------- fc1_w -> f16, N-pad 120->128, K-pad: W16 [128][KP] ----------------
__global__ void w16_kernel(const float* __restrict__ w, _Float16* __restrict__ W16) {
    int idx = blockIdx.x * blockDim.x + threadIdx.x;
    const int total = 128 * KP;
    if (idx >= total) return;
    int k = idx % KP;
    int n = idx / KP;
    float v = (n < 120 && k < KREAL) ? w[(size_t)n * KREAL + k] : 0.0f;
    W16[idx] = (_Float16)v;
}

// ---------------- fc1 GEMM via v_wmma_f32_16x16x32_f16 ----------------
// grid (64 tiles, 16 k-splits), block = 32 (one wave). partial: [KSPLIT][128][128] f32
// 2x-unrolled ping-pong: each buffer is reloaded only AFTER the wmma consuming it,
// so live ranges don't overlap -> no rotate movs; each wmma waits loadcnt<=4.
__global__ __launch_bounds__(32) void fc1_wmma_kernel(const _Float16* __restrict__ A,
                                                      const _Float16* __restrict__ W16,
                                                      float* __restrict__ partial) {
    int tile  = blockIdx.x;          // 0..63
    int split = blockIdx.y;          // 0..15
    int mtile = tile >> 3;
    int ntile = tile & 7;
    int lane  = threadIdx.x;
    int hi    = lane >> 4;           // half-wave
    int r     = lane & 15;

    const int stepsPer = (KSTEPS + KSPLIT - 1) / KSPLIT;   // 110
    int s0 = split * stepsPer;
    int s1 = s0 + stepsPer; if (s1 > KSTEPS) s1 = KSTEPS;  // >= 95 steps per split

    // A fragment (16x32): lane row M=r, contiguous K range hi*16..hi*16+15
    const _Float16* ap = A   + (size_t)(mtile * 16 + r) * KP + hi * 16 + (size_t)s0 * 32;
    // B fragment (32x16): lane col N=r, contiguous K range hi*16..hi*16+15
    const _Float16* bp = W16 + (size_t)(ntile * 16 + r) * KP + hi * 16 + (size_t)s0 * 32;

    f32x8 c = {};

    // prologue: buffer 0 <- step t=s0
    half8 a0l = *(const half8*)(ap);
    half8 a0h = *(const half8*)(ap + 8);
    half8 b0l = *(const half8*)(bp);
    half8 b0h = *(const half8*)(bp + 8);
    half16 a0v = __builtin_shufflevector(a0l, a0h, 0,1,2,3,4,5,6,7,8,9,10,11,12,13,14,15);
    half16 b0v = __builtin_shufflevector(b0l, b0h, 0,1,2,3,4,5,6,7,8,9,10,11,12,13,14,15);
    ap += 32;
    bp += 32;

    int t = s0;
    for (; t + 2 < s1; t += 2) {
        // buffer 1 <- step t+1 (issued before wmma(t))
        half8 na0 = *(const half8*)(ap);
        half8 na1 = *(const half8*)(ap + 8);
        half8 nb0 = *(const half8*)(bp);
        half8 nb1 = *(const half8*)(bp + 8);
        half16 a1v = __builtin_shufflevector(na0, na1, 0,1,2,3,4,5,6,7,8,9,10,11,12,13,14,15);
        half16 b1v = __builtin_shufflevector(nb0, nb1, 0,1,2,3,4,5,6,7,8,9,10,11,12,13,14,15);
        c = __builtin_amdgcn_wmma_f32_16x16x32_f16(false, a0v, false, b0v, (short)0, c,
                                                   false, false);
        // buffer 0 <- step t+2 (a0v/b0v are dead: no overlap, no movs)
        half8 ma0 = *(const half8*)(ap + 32);
        half8 ma1 = *(const half8*)(ap + 40);
        half8 mb0 = *(const half8*)(bp + 32);
        half8 mb1 = *(const half8*)(bp + 40);
        a0v = __builtin_shufflevector(ma0, ma1, 0,1,2,3,4,5,6,7,8,9,10,11,12,13,14,15);
        b0v = __builtin_shufflevector(mb0, mb1, 0,1,2,3,4,5,6,7,8,9,10,11,12,13,14,15);
        c = __builtin_amdgcn_wmma_f32_16x16x32_f16(false, a1v, false, b1v, (short)0, c,
                                                   false, false);
        ap += 64;
        bp += 64;
    }
    // epilogue: 1 or 2 steps remain (t, and possibly t+1)
    if (t + 1 < s1) {
        half8 na0 = *(const half8*)(ap);
        half8 na1 = *(const half8*)(ap + 8);
        half8 nb0 = *(const half8*)(bp);
        half8 nb1 = *(const half8*)(bp + 8);
        half16 a1v = __builtin_shufflevector(na0, na1, 0,1,2,3,4,5,6,7,8,9,10,11,12,13,14,15);
        half16 b1v = __builtin_shufflevector(nb0, nb1, 0,1,2,3,4,5,6,7,8,9,10,11,12,13,14,15);
        c = __builtin_amdgcn_wmma_f32_16x16x32_f16(false, a0v, false, b0v, (short)0, c,
                                                   false, false);
        c = __builtin_amdgcn_wmma_f32_16x16x32_f16(false, a1v, false, b1v, (short)0, c,
                                                   false, false);
    } else {
        c = __builtin_amdgcn_wmma_f32_16x16x32_f16(false, a0v, false, b0v, (short)0, c,
                                                   false, false);
    }

    // C/D: VGPR rr -> M = rr + hi*8, N = r
    float* outp = partial + ((size_t)split * 128 + (size_t)mtile * 16) * 128 + ntile * 16;
    #pragma unroll
    for (int rr = 0; rr < 8; ++rr) {
        outp[(hi * 8 + rr) * 128 + r] = c[rr];
    }
}

// ---------------- reduce K-splits + bias + relu -> h3 [128][120] ----------------
__global__ void fc1_reduce_kernel(const float* __restrict__ partial,
                                  const float* __restrict__ bias, float* __restrict__ h3) {
    int idx = blockIdx.x * blockDim.x + threadIdx.x;
    if (idx >= B_ * 120) return;
    int n = idx % 120;
    int m = idx / 120;
    float acc = bias[n];
    #pragma unroll
    for (int s = 0; s < KSPLIT; ++s)
        acc += partial[((size_t)s * 128 + m) * 128 + n];
    h3[idx] = fmaxf(acc, 0.0f);
}

// ---------------- fc2 + relu: [128][120] x [84][120]^T -> [128][84] ----------------
__global__ void fc2_kernel(const float* __restrict__ h3, const float* __restrict__ w,
                           const float* __restrict__ bias, float* __restrict__ h4) {
    int idx = blockIdx.x * blockDim.x + threadIdx.x;
    if (idx >= B_ * 84) return;
    int j = idx % 84;
    int m = idx / 84;
    float acc = bias[j];
    const float* hr = h3 + m * 120;
    const float* wr = w + j * 120;
    for (int i = 0; i < 120; ++i) acc = fmaf(hr[i], wr[i], acc);
    h4[idx] = fmaxf(acc, 0.0f);
}

// ---------------- fc3: [128][84] x [1][84]^T -> s [128] ----------------
__global__ void fc3_kernel(const float* __restrict__ h4, const float* __restrict__ w,
                           const float* __restrict__ bias, float* __restrict__ s) {
    int m = blockIdx.x * blockDim.x + threadIdx.x;
    if (m >= B_) return;
    float acc = bias[0];
    const float* hr = h4 + m * 84;
    for (int j = 0; j < 84; ++j) acc = fmaf(hr[j], w[j], acc);
    s[m] = acc;
}

// ---------------- fidelity-graph aggregation + tanh/sigmoid head ----------------
__global__ void head_kernel(const float* __restrict__ s, const float* __restrict__ ph_w,
                            const float* __restrict__ ph_b, const float* __restrict__ ph_scale,
                            const float* __restrict__ ph_shift, float* __restrict__ out) {
    __shared__ float sv[B_];
    int i = threadIdx.x;
    sv[i] = s[i];
    __syncthreads();
    float si  = sv[i];
    float sni = si / (fabsf(si) + 1e-12f);
    float agg = 0.0f, deg = 0.0f;
    for (int j = 0; j < B_; ++j) {
        if (j == i) continue;
        float sj  = sv[j];
        float snj = sj / (fabsf(sj) + 1e-12f);
        float fid = sni * snj;
        fid *= fid;
        if (fid >= 0.8f) { agg += sj; deg += 1.0f; }
    }
    float g = (deg > 0.0f) ? agg : si;
    float p = tanhf(g * ph_w[0] + ph_b[0]) * ph_scale[0] + ph_shift[0];
    out[i] = 1.0f / (1.0f + expf(-p));
}

extern "C" void kernel_launch(void* const* d_in, const int* in_sizes, int n_in,
                              void* d_out, int out_size, void* d_ws, size_t ws_size,
                              hipStream_t stream) {
    const float* x       = (const float*)d_in[0];
    const float* conv1_w = (const float*)d_in[1];
    const float* conv1_b = (const float*)d_in[2];
    const float* conv2_w = (const float*)d_in[3];
    const float* conv2_b = (const float*)d_in[4];
    const float* fc1_w   = (const float*)d_in[5];
    const float* fc1_b   = (const float*)d_in[6];
    const float* fc2_w   = (const float*)d_in[7];
    const float* fc2_b   = (const float*)d_in[8];
    const float* fc3_w   = (const float*)d_in[9];
    const float* fc3_b   = (const float*)d_in[10];
    const float* ph_w    = (const float*)d_in[11];
    const float* ph_b    = (const float*)d_in[12];
    const float* ph_sc   = (const float*)d_in[13];
    const float* ph_sh   = (const float*)d_in[14];
    float* out = (float*)d_out;
    char* ws = (char*)d_ws;

    // Workspace layout (bytes), with reuse:
    //   [0)            h1 (47,235,072)  -> later h2 (29,521,920) -> later partial (1,048,576)
    //   [47,235,072)   p1 (46,476,288)  -> later A f16 (14,295,040) + W16 f16 (14,295,040)
    //   [small tail]   h3 / h4 / s
    const size_t off_h1   = 0;
    const size_t off_p1   = 47235072;                    // 128*6*124*124*4, 256-aligned
    const size_t off_h2   = 0;                           // reuse h1 space
    const size_t off_A    = off_p1;                      // reuse p1 space
    const size_t off_W16  = off_p1 + (size_t)B_ * KP * 2;  // +14,295,040
    const size_t off_part = 0;                           // reuse h2 space
    const size_t off_h3   = (size_t)KSPLIT * 128 * 128 * 4;      // 1,048,576
    const size_t off_h4   = off_h3 + (size_t)B_ * 120 * 4;
    const size_t off_s    = off_h4 + (size_t)B_ * 84 * 4;
    (void)ws_size;

    float*     h1   = (float*)(ws + off_h1);
    float*     p1   = (float*)(ws + off_p1);
    float*     h2   = (float*)(ws + off_h2);
    _Float16*  A    = (_Float16*)(ws + off_A);
    _Float16*  W16  = (_Float16*)(ws + off_W16);
    float*     part = (float*)(ws + off_part);
    float*     h3   = (float*)(ws + off_h3);
    float*     h4   = (float*)(ws + off_h4);
    float*     sbuf = (float*)(ws + off_s);

    const int T = 256;
    int n1 = B_ * 6 * 124 * 124;
    conv1_kernel<<<ceil_div(n1, T), T, 0, stream>>>(x, conv1_w, conv1_b, h1);

    int np1 = B_ * 6 * 123 * 123;
    maxpool_kernel<<<ceil_div(np1, T), T, 0, stream>>>(h1, p1, 6, 124, 124, np1);

    int n2 = B_ * 15 * 62 * 62;
    conv2_kernel<<<ceil_div(n2, T), T, 0, stream>>>(p1, conv2_w, conv2_b, h2);

    int nA = B_ * KP;
    pool2_f16_kernel<<<ceil_div(nA, T), T, 0, stream>>>(h2, A);

    int nW = 128 * KP;
    w16_kernel<<<ceil_div(nW, T), T, 0, stream>>>(fc1_w, W16);

    fc1_wmma_kernel<<<dim3(64, KSPLIT, 1), dim3(32, 1, 1), 0, stream>>>(A, W16, part);

    fc1_reduce_kernel<<<ceil_div(B_ * 120, T), T, 0, stream>>>(part, fc1_b, h3);

    fc2_kernel<<<ceil_div(B_ * 84, T), T, 0, stream>>>(h3, fc2_w, fc2_b, h4);

    fc3_kernel<<<1, B_, 0, stream>>>(h4, fc3_w, fc3_b, sbuf);

    head_kernel<<<1, B_, 0, stream>>>(sbuf, ph_w, ph_b, ph_sc, ph_sh, out);
}
